// PromptGuidedRoutingAttention_2336462209423
// MI455X (gfx1250) — matrix operand order repaired and checked by reading, not compiled, not run on hardware
//
#include <hip/hip_runtime.h>

typedef __attribute__((ext_vector_type(16))) __bf16 v16bf;
typedef __attribute__((ext_vector_type(8)))  float  v8f;
typedef __attribute__((ext_vector_type(4)))  unsigned int uv4;

#define DEV static __device__ __forceinline__

DEV unsigned short f2bf(float f) {
  unsigned int u = __float_as_uint(f);
  u = (u + 0x7FFFu + ((u >> 16) & 1u)) >> 16;   // RNE
  return (unsigned short)u;
}
DEV float bf2f(unsigned short h) { return __uint_as_float(((unsigned int)h) << 16); }

DEV v8f wmma_bf16(v16bf a, v16bf b, v8f c) {
  return __builtin_amdgcn_wmma_f32_16x16x32_bf16(false, a, false, b, (short)0, c, false, false);
}

// ---- DPP16 16-lane butterfly reductions (no LDS round-trips) ----
// xor1 = quad_perm[1,0,3,2] (0xB1), xor2 = quad_perm[2,3,0,1] (0x4E),
// xor7 = row_half_mirror (0x141), xor15 = row_mirror (0x140).
template <int CTRL>
DEV float dpp_mov(float x) {
  return __int_as_float(
      __builtin_amdgcn_update_dpp(0, __float_as_int(x), CTRL, 0xF, 0xF, true));
}
DEV float red_max16(float x) {
  x = fmaxf(x, dpp_mov<0xB1>(x));
  x = fmaxf(x, dpp_mov<0x4E>(x));
  x = fmaxf(x, dpp_mov<0x141>(x));
  x = fmaxf(x, dpp_mov<0x140>(x));
  return x;
}
DEV float red_sum16(float x) {
  x += dpp_mov<0xB1>(x);
  x += dpp_mov<0x4E>(x);
  x += dpp_mov<0x141>(x);
  x += dpp_mov<0x140>(x);
  return x;
}

// A-fragment (16x32, M x K) from row-major [row][col] bf16 (ushort) buffer.
// Lane L: M = L%16; K contiguous 16 starting at k0 (+16 for high half).
DEV v16bf load_frag_a(const unsigned short* base, int row0, int ld, int k0) {
  int lane = threadIdx.x & 31;
  const unsigned short* p = base + (size_t)(row0 + (lane & 15)) * ld + k0 + ((lane & 16) ? 16 : 0);
  union { v16bf v; uv4 q[2]; } u;
  u.q[0] = *(const uv4*)(p);
  u.q[1] = *(const uv4*)(p + 8);
  return u.v;
}

// B-fragment (32x16, K x N) where memory stores B^T row-major: mem[n][k]
// (K-matrix [key][channel], weights W[o][c], or V^T [dim][token]).
// Lane L: N = L%16; K contiguous 16 starting at k0 (+16 for high half).
DEV v16bf load_frag_bt(const unsigned short* base, int n0, int ld, int k0) {
  int lane = threadIdx.x & 31;
  const unsigned short* p = base + (size_t)(n0 + (lane & 15)) * ld + k0 + ((lane & 16) ? 16 : 0);
  union { v16bf v; uv4 q[2]; } u;
  u.q[0] = *(const uv4*)(p);
  u.q[1] = *(const uv4*)(p + 8);
  return u.v;
}

// ---------------------------------------------------------------- elementwise
__global__ void k_f2bf(const float* __restrict__ in, unsigned short* __restrict__ out, int n) {
  int i = blockIdx.x * blockDim.x + threadIdx.x;
  if (i < n) out[i] = f2bf(in[i]);
}

// [b,c,64,64] f32 -> window tokens [b, n(8x8 windows), t(64), c] bf16
__global__ void k_win_tokens(const float* __restrict__ in, unsigned short* __restrict__ out, int total) {
  int i = blockIdx.x * blockDim.x + threadIdx.x;
  if (i >= total) return;
  int xx = i & 63, y = (i >> 6) & 63, c = (i >> 12) & 255, b = i >> 20;
  int n = ((y >> 3) << 3) + (xx >> 3);
  int t = ((y & 7) << 3) + (xx & 7);
  out[((((size_t)b * 64 + n) * 64 + t) << 8) + c] = f2bf(in[i]);
}

// window mean descriptor: [b*nw, 256] f32 from bf16 tokens
__global__ void __launch_bounds__(256) k_desc(const unsigned short* __restrict__ tok, float* __restrict__ desc) {
  int wid = blockIdx.x;
  int c = threadIdx.x;
  const unsigned short* p = tok + (size_t)wid * (64 * 256) + c;
  float s = 0.f;
#pragma unroll 8
  for (int t = 0; t < 64; ++t) s += bf2f(p[t * 256]);
  desc[(size_t)wid * 256 + c] = s * (1.0f / 64.0f);
}

// routing: per (b, qwin), dot with 64 prompt descriptors, top-4 (first index wins ties)
__global__ void __launch_bounds__(64) k_route(const float* __restrict__ xd, const float* __restrict__ pd,
                                              int* __restrict__ routed) {
  __shared__ float sc[64];
  int b = blockIdx.x >> 6;
  int k = threadIdx.x;
  const float* a = xd + (size_t)blockIdx.x * 256;
  const float* pv = pd + ((size_t)b * 64 + k) * 256;
  float s = 0.f;
  for (int i = 0; i < 256; ++i) s += a[i] * pv[i];
  sc[k] = s;
  __syncthreads();
  if (k == 0) {
    int* r = routed + blockIdx.x * 4;
    for (int j = 0; j < 4; ++j) {
      float best = -3.0e38f; int bi = 0;
      for (int m = 0; m < 64; ++m) if (sc[m] > best) { best = sc[m]; bi = m; }
      r[j] = bi;
      sc[bi] = -3.0e38f;
    }
  }
}

// ------------------------------------------------------------------- GEMM
// out[row, 0..255] = A[row, 0..K-1] @ W[256, K]^T   (A, W bf16)
// grid.x = rows/64, block = 256 (8 waves, each owns a 32-col slice).
// A panel (64 x K) staged once through LDS (coalesced b128), then ds_load frags.
// OUT_MODE: 0 = bf16 row-major, 1 = f32 row-major,
//           2 = bf16 window-transposed [win][col][t]  (for V)
template <int K, int OUT_MODE>
__global__ void __launch_bounds__(256) k_gemm(const unsigned short* __restrict__ A,
                                              const unsigned short* __restrict__ W,
                                              void* __restrict__ out) {
  __shared__ unsigned short aLds[64 * K];
  int row0 = blockIdx.x * 64;
  int wave = threadIdx.x >> 5;
  int lane = threadIdx.x & 31;
  int nbase = wave * 32;

  // cooperative stage of A panel: 64*K ushorts as b128 chunks
  {
    const uv4* src = (const uv4*)(A + (size_t)row0 * K);
    uv4* dst = (uv4*)aLds;
    constexpr int CHUNKS = 64 * K / 8;
#pragma unroll
    for (int i = 0; i < CHUNKS / 256; ++i)
      dst[threadIdx.x + i * 256] = src[threadIdx.x + i * 256];
  }
  __syncthreads();

  v8f z = {};
  v8f acc[4][2];
#pragma unroll
  for (int mt = 0; mt < 4; ++mt) { acc[mt][0] = z; acc[mt][1] = z; }

  for (int ks = 0; ks < K / 32; ++ks) {
    int k0 = ks * 32;
    v16bf a[4];
#pragma unroll
    for (int mt = 0; mt < 4; ++mt) a[mt] = load_frag_a(aLds, mt * 16, K, k0);
#pragma unroll
    for (int nb = 0; nb < 2; ++nb) {
      v16bf b = load_frag_bt(W, nbase + nb * 16, K, k0);
#pragma unroll
      for (int mt = 0; mt < 4; ++mt) acc[mt][nb] = wmma_bf16(a[mt], b, acc[mt][nb]);
    }
  }

  int hi8 = (lane & 16) ? 8 : 0;
  int nsel = lane & 15;
#pragma unroll
  for (int mt = 0; mt < 4; ++mt)
#pragma unroll
    for (int nb = 0; nb < 2; ++nb)
#pragma unroll
      for (int r = 0; r < 8; ++r) {
        int row = row0 + mt * 16 + r + hi8;
        int col = nbase + nb * 16 + nsel;
        if constexpr (OUT_MODE == 0) {
          ((unsigned short*)out)[(size_t)row * 256 + col] = f2bf(acc[mt][nb][r]);
        } else if constexpr (OUT_MODE == 1) {
          ((float*)out)[(size_t)row * 256 + col] = acc[mt][nb][r];
        } else {
          size_t win = (size_t)(row >> 6);
          int t = row & 63;
          ((unsigned short*)out)[win * (256 * 64) + (size_t)col * 64 + t] = f2bf(acc[mt][nb][r]);
        }
      }
}

// ------------------------------------------------------------------ attention
// 1 block = (batch, query window); wave w = head w. Flash softmax over 4 routed
// prompt windows (64 keys each). Q,K bf16 [win][token][ch]; V bf16 transposed
// [win][ch][token] so V fragments are K-contiguous b128 loads.
__global__ void __launch_bounds__(256) k_attn(const unsigned short* __restrict__ Qb,
                                              const unsigned short* __restrict__ Kb,
                                              const unsigned short* __restrict__ Vt,
                                              const int* __restrict__ routed,
                                              unsigned short* __restrict__ Ob) {
  __shared__ unsigned short pLds[8][64 * 64];   // 64 KB: per-wave P tile (C->A relayout)
  int blk = blockIdx.x;
  int head = threadIdx.x >> 5;
  int lane = threadIdx.x & 31;
  int nsel = lane & 15;
  int hi8 = (lane & 16) ? 8 : 0;
  int ch0 = head * 32;

  const unsigned short* Qw = Qb + (size_t)blk * (64 * 256);
  v16bf qa[4];
#pragma unroll
  for (int mt = 0; mt < 4; ++mt) qa[mt] = load_frag_a(Qw, mt * 16, 256, ch0);

  v8f z = {};
  v8f o[4][2];
  float mrow[4][8], lrow[4][8];
#pragma unroll
  for (int mt = 0; mt < 4; ++mt) {
    o[mt][0] = z; o[mt][1] = z;
#pragma unroll
    for (int r = 0; r < 8; ++r) { mrow[mt][r] = -3.0e38f; lrow[mt][r] = 0.f; }
  }

  const int* ridx = routed + blk * 4;
  size_t batchBase = (size_t)(blk >> 6) * 64 * (64 * 256);
  const float scale = 0.17677669529663687f;      // 1/sqrt(32)
  unsigned short* myP = &pLds[head][0];

  for (int j = 0; j < 4; ++j) {
    int kw = ridx[j];
    const unsigned short* Kw = Kb + batchBase + (size_t)kw * (64 * 256);
    const unsigned short* Vw = Vt + batchBase + (size_t)kw * (64 * 256);

    v16bf kfr[4];
#pragma unroll
    for (int nt = 0; nt < 4; ++nt) kfr[nt] = load_frag_bt(Kw, nt * 16, 256, ch0);

#pragma unroll
    for (int mt = 0; mt < 4; ++mt) {
      v8f s[4];
#pragma unroll
      for (int nt = 0; nt < 4; ++nt) s[nt] = wmma_bf16(qa[mt], kfr[nt], z);
#pragma unroll
      for (int r = 0; r < 8; ++r) {
        float s0 = s[0][r] * scale, s1 = s[1][r] * scale;
        float s2 = s[2][r] * scale, s3 = s[3][r] * scale;
        float cm = red_max16(fmaxf(fmaxf(s0, s1), fmaxf(s2, s3)));
        float nm = fmaxf(mrow[mt][r], cm);
        float alpha = __expf(mrow[mt][r] - nm);
        mrow[mt][r] = nm;
        float p0 = __expf(s0 - nm), p1 = __expf(s1 - nm);
        float p2 = __expf(s2 - nm), p3 = __expf(s3 - nm);
        float rs = red_sum16(p0 + p1 + p2 + p3);
        lrow[mt][r] = lrow[mt][r] * alpha + rs;
        o[mt][0][r] *= alpha;
        o[mt][1][r] *= alpha;
        int row = (mt * 16 + r + hi8) * 64;          // wave-private region: no barrier needed
        myP[row + nsel]      = f2bf(p0);
        myP[row + 16 + nsel] = f2bf(p1);
        myP[row + 32 + nsel] = f2bf(p2);
        myP[row + 48 + nsel] = f2bf(p3);
      }
    }

#pragma unroll
    for (int ks = 0; ks < 2; ++ks) {
      v16bf pa[4];
#pragma unroll
      for (int mt = 0; mt < 4; ++mt) pa[mt] = load_frag_a(myP, mt * 16, 64, ks * 32);
#pragma unroll
      for (int nb = 0; nb < 2; ++nb) {
        // V^T layout [dim][token]: B[K=token][N=dim] is K-contiguous -> b128 loads
        v16bf vfr = load_frag_bt(Vw, ch0 + nb * 16, 64, ks * 32);
#pragma unroll
        for (int mt = 0; mt < 4; ++mt) o[mt][nb] = wmma_bf16(pa[mt], vfr, o[mt][nb]);
      }
    }
  }

  unsigned short* Ow = Ob + (size_t)blk * (64 * 256);
#pragma unroll
  for (int mt = 0; mt < 4; ++mt)
#pragma unroll
    for (int nb = 0; nb < 2; ++nb)
#pragma unroll
      for (int r = 0; r < 8; ++r) {
        float v = o[mt][nb][r] / lrow[mt][r];
        Ow[(size_t)(mt * 16 + r + hi8) * 256 + ch0 + nb * 16 + nsel] = f2bf(v);
      }
}

// ----------------------------------------------------------------- gate path
__global__ void k_pack_x(const float* __restrict__ x, unsigned short* __restrict__ cat, int total) {
  int i = blockIdx.x * blockDim.x + threadIdx.x;
  if (i >= total) return;
  int xx = i & 63, y = (i >> 6) & 63, c = (i >> 12) & 255, b = i >> 20;
  size_t p = ((size_t)b << 12) + (y << 6) + xx;
  cat[p * 512 + c] = f2bf(x[i]);
}

__global__ void k_pack_y(const float* __restrict__ ytok, unsigned short* __restrict__ cat, int total) {
  int i = blockIdx.x * blockDim.x + threadIdx.x;   // (pixel, c) output-major
  if (i >= total) return;
  int c = i & 255, p = i >> 8;
  int b = p >> 12, y = (p >> 6) & 63, xx = p & 63;
  int n = ((y >> 3) << 3) + (xx >> 3);
  int t = ((y & 7) << 3) + (xx & 7);
  cat[(size_t)p * 512 + 256 + c] = f2bf(ytok[((((size_t)b * 64 + n) * 64 + t) << 8) + c]);
}

__global__ void __launch_bounds__(256) k_bn_stats(const float* __restrict__ g, float* __restrict__ stats) {
  int o = blockIdx.x;
  __shared__ float ss[256], s2[256];
  float a = 0.f, b2 = 0.f;
  for (int p = threadIdx.x; p < 32768; p += 256) {
    float v = g[(size_t)p * 256 + o];
    a += v; b2 += v * v;
  }
  ss[threadIdx.x] = a; s2[threadIdx.x] = b2;
  __syncthreads();
  for (int s = 128; s > 0; s >>= 1) {
    if (threadIdx.x < s) { ss[threadIdx.x] += ss[threadIdx.x + s]; s2[threadIdx.x] += s2[threadIdx.x + s]; }
    __syncthreads();
  }
  if (threadIdx.x == 0) {
    float mean = ss[0] * (1.0f / 32768.0f);
    float var = s2[0] * (1.0f / 32768.0f) - mean * mean;
    stats[o] = mean; stats[256 + o] = var;
  }
}

__global__ void k_final(const float* __restrict__ x, const float* __restrict__ ytok,
                        const float* __restrict__ g, const float* __restrict__ stats,
                        const float* __restrict__ gamma, const float* __restrict__ beta,
                        float* __restrict__ out, int total) {
  int i = blockIdx.x * blockDim.x + threadIdx.x;   // [b][c][y][x]
  if (i >= total) return;
  int xx = i & 63, y = (i >> 6) & 63, c = (i >> 12) & 255, b = i >> 20;
  size_t p = ((size_t)b << 12) + (y << 6) + xx;
  int n = ((y >> 3) << 3) + (xx >> 3);
  int t = ((y & 7) << 3) + (xx & 7);
  float gv = g[p * 256 + c];
  float gn = (gv - stats[c]) * rsqrtf(stats[256 + c] + 1e-5f) * gamma[c] + beta[c];
  float gate = 1.0f / (1.0f + __expf(-gn));
  float yv = ytok[((((size_t)b * 64 + n) * 64 + t) << 8) + c];
  out[i] = x[i] + gate * yv;
}

// -------------------------------------------------------------- launch glue
extern "C" void kernel_launch(void* const* d_in, const int* in_sizes, int n_in,
                              void* d_out, int out_size, void* d_ws, size_t ws_size,
                              hipStream_t stream) {
  const float* x      = (const float*)d_in[0];
  const float* prompt = (const float*)d_in[1];
  const float* Wq     = (const float*)d_in[2];
  const float* Wk     = (const float*)d_in[3];
  const float* Wv     = (const float*)d_in[4];
  const float* Wproj  = (const float*)d_in[5];
  const float* Wg     = (const float*)d_in[6];
  const float* gamma  = (const float*)d_in[7];
  const float* beta   = (const float*)d_in[8];
  float* out = (float*)d_out;

  constexpr size_t SZ_W    = 256 * 256;                 // weight elems
  constexpr size_t SZ_WG   = 256 * 512;
  constexpr size_t SZ_TOK  = (size_t)8 * 64 * 64 * 256; // 8388608 tokens*ch
  constexpr size_t SZ_DESC = (size_t)8 * 64 * 256;
  constexpr size_t SZ_RT   = (size_t)8 * 64 * 4;
  constexpr size_t SZ_PIXC = (size_t)32768 * 256;
  constexpr size_t SZ_CAT  = (size_t)32768 * 512;

  constexpr size_t OFF_WQ   = 0;
  constexpr size_t OFF_WK   = OFF_WQ   + SZ_W * 2;
  constexpr size_t OFF_WV   = OFF_WK   + SZ_W * 2;
  constexpr size_t OFF_WP   = OFF_WV   + SZ_W * 2;
  constexpr size_t OFF_WG   = OFF_WP   + SZ_W * 2;
  constexpr size_t OFF_XTOK = OFF_WG   + SZ_WG * 2;
  constexpr size_t OFF_PTOK = OFF_XTOK + SZ_TOK * 2;
  constexpr size_t OFF_XD   = OFF_PTOK + SZ_TOK * 2;
  constexpr size_t OFF_PD   = OFF_XD   + SZ_DESC * 4;
  constexpr size_t OFF_RT   = OFF_PD   + SZ_DESC * 4;
  constexpr size_t OFF_Q    = OFF_RT   + SZ_RT * 4;
  constexpr size_t OFF_K    = OFF_Q    + SZ_TOK * 2;
  constexpr size_t OFF_V    = OFF_K    + SZ_TOK * 2;
  constexpr size_t OFF_AO   = OFF_V    + SZ_TOK * 2;
  constexpr size_t OFF_YT   = OFF_AO   + SZ_TOK * 2;
  constexpr size_t OFF_CAT  = OFF_YT   + SZ_PIXC * 4;
  constexpr size_t OFF_G    = OFF_CAT  + SZ_CAT * 2;
  constexpr size_t OFF_BN   = OFF_G    + SZ_PIXC * 4;

  char* ws = (char*)d_ws;
  unsigned short* wq_bf = (unsigned short*)(ws + OFF_WQ);
  unsigned short* wk_bf = (unsigned short*)(ws + OFF_WK);
  unsigned short* wv_bf = (unsigned short*)(ws + OFF_WV);
  unsigned short* wp_bf = (unsigned short*)(ws + OFF_WP);
  unsigned short* wg_bf = (unsigned short*)(ws + OFF_WG);
  unsigned short* xtok  = (unsigned short*)(ws + OFF_XTOK);
  unsigned short* ptok  = (unsigned short*)(ws + OFF_PTOK);
  float*          xd    = (float*)(ws + OFF_XD);
  float*          pd    = (float*)(ws + OFF_PD);
  int*            rt    = (int*)(ws + OFF_RT);
  unsigned short* Qb    = (unsigned short*)(ws + OFF_Q);
  unsigned short* Kb    = (unsigned short*)(ws + OFF_K);
  unsigned short* Vt    = (unsigned short*)(ws + OFF_V);
  unsigned short* AO    = (unsigned short*)(ws + OFF_AO);
  float*          YT    = (float*)(ws + OFF_YT);
  unsigned short* CAT   = (unsigned short*)(ws + OFF_CAT);
  float*          G     = (float*)(ws + OFF_G);
  float*          BN    = (float*)(ws + OFF_BN);

  const int TOTAL = 8 * 256 * 64 * 64;          // 8388608
  const int EB = 256;

  // 1) weights -> bf16
  k_f2bf<<<(int)(SZ_W + EB - 1) / EB, EB, 0, stream>>>(Wq, wq_bf, (int)SZ_W);
  k_f2bf<<<(int)(SZ_W + EB - 1) / EB, EB, 0, stream>>>(Wk, wk_bf, (int)SZ_W);
  k_f2bf<<<(int)(SZ_W + EB - 1) / EB, EB, 0, stream>>>(Wv, wv_bf, (int)SZ_W);
  k_f2bf<<<(int)(SZ_W + EB - 1) / EB, EB, 0, stream>>>(Wproj, wp_bf, (int)SZ_W);
  k_f2bf<<<(int)(SZ_WG + EB - 1) / EB, EB, 0, stream>>>(Wg, wg_bf, (int)SZ_WG);

  // 2) window tokens
  k_win_tokens<<<TOTAL / EB, EB, 0, stream>>>(x, xtok, TOTAL);
  k_win_tokens<<<TOTAL / EB, EB, 0, stream>>>(prompt, ptok, TOTAL);

  // 3) descriptors + routing
  k_desc<<<512, 256, 0, stream>>>(xtok, xd);
  k_desc<<<512, 256, 0, stream>>>(ptok, pd);
  k_route<<<512, 64, 0, stream>>>(xd, pd, rt);

  // 4) Q/K/V projections (K/V over ALL prompt windows once; V stored transposed)
  k_gemm<256, 0><<<512, 256, 0, stream>>>(xtok, wq_bf, Qb);
  k_gemm<256, 0><<<512, 256, 0, stream>>>(ptok, wk_bf, Kb);
  k_gemm<256, 2><<<512, 256, 0, stream>>>(ptok, wv_bf, Vt);

  // 5) routed flash attention
  k_attn<<<512, 256, 0, stream>>>(Qb, Kb, Vt, rt, AO);

  // 6) output projection (f32 tokens)
  k_gemm<256, 1><<<512, 256, 0, stream>>>(AO, wp_bf, YT);

  // 7) gate: concat [x, y] (bf16), 1x1 conv via GEMM (K=512), BN stats, fused epilogue
  k_pack_x<<<TOTAL / EB, EB, 0, stream>>>(x, CAT, TOTAL);
  k_pack_y<<<TOTAL / EB, EB, 0, stream>>>(YT, CAT, TOTAL);
  k_gemm<512, 1><<<512, 256, 0, stream>>>(CAT, wg_bf, G);
  k_bn_stats<<<256, 256, 0, stream>>>(G, BN);
  k_final<<<TOTAL / EB, EB, 0, stream>>>(x, YT, G, BN, gamma, beta, out, TOTAL);

  (void)in_sizes; (void)n_in; (void)out_size; (void)ws_size;
}
